// GCN_13099650253144
// MI455X (gfx1250) — matrix operand order repaired and checked by reading, not compiled, not run on hardware
//
#include <hip/hip_runtime.h>
#include <hip/hip_bf16.h>

typedef __attribute__((ext_vector_type(16))) _Float16 v16h;
typedef __attribute__((ext_vector_type(8)))  _Float16 h8;
typedef __attribute__((ext_vector_type(8)))  float    v8f;

#define LDA 40  // padded LDS row stride in halves (80B, multiple of 16B)

// ---------------------------------------------------------------------------
// WMMA GEMM: C[M,N] = A[M,K] * B[K,N], fp32 in/out, f16 multiply, f32 accum.
// Block tile 128x64, BK=32. 8 waves, each wave owns a 32x32 output (2x2 WMMA).
// Software-pipelined: next K-chunk is prefetched into registers while the
// current chunk's WMMAs run; A-row guard replaced by branchless clamp (rows
// >= M compute garbage but are never stored).
// ---------------------------------------------------------------------------
__global__ __launch_bounds__(256)
void wmma_gemm_f16(const float* __restrict__ A, const float* __restrict__ Bm,
                   float* __restrict__ C, int M, int K, int N)
{
    __shared__ _Float16 la[128 * LDA];   // A tile [128 rows][32 k], row-major
    __shared__ _Float16 lb[64  * LDA];   // B tile stored as [n][k]

    const int tid  = threadIdx.x;
    const int lane = tid & 31;
    const int wv   = tid >> 5;
    const int wm   = wv & 3;        // row group 0..3 (32 rows each)
    const int wn   = wv >> 2;       // col group 0..1 (32 cols each)
    const int hf   = lane >> 4;     // lane half (0: lanes 0-15, 1: lanes 16-31)
    const int l16  = lane & 15;
    const int bm   = blockIdx.x * 128;
    const int bn   = blockIdx.y * 64;

    // Per-thread staging coordinates (fixed across K-chunks).
    const int ar[4] = { (tid + 0*256) >> 3, (tid + 1*256) >> 3,
                        (tid + 2*256) >> 3, (tid + 3*256) >> 3 };
    const int ac    = (tid & 7) << 2;                 // float4 col within 32-K
    const int bk[2] = { (tid + 0*256) >> 4, (tid + 1*256) >> 4 };
    const int bn4   = (tid & 15) << 2;

    v8f acc[2][2] = {};
    float4 ra[4], rb[2];

    // --- initial tile fetch (k0 = 0), branchless row clamp ---
    #pragma unroll
    for (int i = 0; i < 4; ++i) {
        int gr = min(bm + ar[i], M - 1);
        ra[i] = *(const float4*)(A + (size_t)gr * K + ac);
    }
    #pragma unroll
    for (int i = 0; i < 2; ++i)
        rb[i] = *(const float4*)(Bm + (size_t)bk[i] * N + bn + bn4);

    for (int k0 = 0; k0 < K; k0 += 32) {
        // regs -> LDS (f32 -> f16 convert)
        #pragma unroll
        for (int i = 0; i < 4; ++i) {
            _Float16* d = &la[ar[i] * LDA + ac];
            d[0] = (_Float16)ra[i].x; d[1] = (_Float16)ra[i].y;
            d[2] = (_Float16)ra[i].z; d[3] = (_Float16)ra[i].w;
        }
        #pragma unroll
        for (int i = 0; i < 2; ++i) {
            lb[(bn4 + 0) * LDA + bk[i]] = (_Float16)rb[i].x;
            lb[(bn4 + 1) * LDA + bk[i]] = (_Float16)rb[i].y;
            lb[(bn4 + 2) * LDA + bk[i]] = (_Float16)rb[i].z;
            lb[(bn4 + 3) * LDA + bk[i]] = (_Float16)rb[i].w;
        }
        __syncthreads();

        // prefetch next K-chunk into registers (uniform branch, no divergence)
        const int kn = k0 + 32;
        if (kn < K) {
            #pragma unroll
            for (int i = 0; i < 4; ++i) {
                int gr = min(bm + ar[i], M - 1);
                ra[i] = *(const float4*)(A + (size_t)gr * K + kn + ac);
            }
            #pragma unroll
            for (int i = 0; i < 2; ++i)
                rb[i] = *(const float4*)(Bm + (size_t)(kn + bk[i]) * N + bn + bn4);
        }

        // Fragment loads per documented gfx1250 layouts.
        // A 16x32 f16: lanes 0-15 hold K {0..7,16..23}, lanes 16-31 K {8..15,24..31}.
        // B 32x16 f16: lanes 0-15 hold K {0..7,16..23}, lanes 16-31 K {8..15,24..31}, N = lane%16.
        v16h af[2], bf[2];
        #pragma unroll
        for (int s = 0; s < 2; ++s) {
            int m = wm * 32 + s * 16 + l16;
            h8 alo = *(const h8*)&la[m * LDA + hf * 8];
            h8 ahi = *(const h8*)&la[m * LDA + 16 + hf * 8];
            af[s] = __builtin_shufflevector(alo, ahi,
                     0,1,2,3,4,5,6,7,8,9,10,11,12,13,14,15);
            int nn = wn * 32 + s * 16 + l16;
            h8 blo = *(const h8*)&lb[nn * LDA + hf * 8];
            h8 bhi = *(const h8*)&lb[nn * LDA + 16 + hf * 8];
            bf[s] = __builtin_shufflevector(blo, bhi,
                     0,1,2,3,4,5,6,7,8,9,10,11,12,13,14,15);
        }
        #pragma unroll
        for (int sm = 0; sm < 2; ++sm)
            #pragma unroll
            for (int sn = 0; sn < 2; ++sn)
                acc[sm][sn] = __builtin_amdgcn_wmma_f32_16x16x32_f16(
                    false, af[sm], false, bf[sn],
                    (short)0, acc[sm][sn], false, false);
        __syncthreads();
    }

    // C/D layout: VGPR r -> M = base + hf*8 + r; N = base + lane%16.
    #pragma unroll
    for (int sm = 0; sm < 2; ++sm) {
        #pragma unroll
        for (int sn = 0; sn < 2; ++sn) {
            int gcol = bn + wn * 32 + sn * 16 + l16;
            #pragma unroll
            for (int r = 0; r < 8; ++r) {
                int grow = bm + wm * 32 + sm * 16 + hf * 8 + r;
                if (grow < M) C[(size_t)grow * N + gcol] = acc[sm][sn][r];
            }
        }
    }
}

// ---------------------------------------------------------------------------
// Degree / normalization
// ---------------------------------------------------------------------------
__global__ void deg_init(float* deg, int n) {
    int i = blockIdx.x * blockDim.x + threadIdx.x;
    if (i < n) deg[i] = 1.0f;                       // self-loop contribution
}

__global__ void deg_count(const long long* __restrict__ cols, float* deg, long long E) {
    long long e = (long long)blockIdx.x * blockDim.x + threadIdx.x;
    if (e < E) atomicAdd(&deg[cols[e]], 1.0f);
}

__global__ void deg_rsqrt(const float* __restrict__ deg, float* dinv, int n) {
    int i = blockIdx.x * blockDim.x + threadIdx.x;
    if (i < n) dinv[i] = rsqrtf(deg[i]);
}

// agg[i,:] = xw[i,:] * dinv[i]^2   (self-loop term; also initializes agg)
__global__ void self_init(const float* __restrict__ xw, const float* __restrict__ dinv,
                          float* __restrict__ agg, int n, int D) {
    size_t i = (size_t)blockIdx.x * blockDim.x + threadIdx.x;
    size_t total = (size_t)n * D;
    if (i >= total) return;
    float di = dinv[i / D];
    agg[i] = xw[i] * di * di;
}

// One wave per edge: agg[col,:] += xw[row,:] * dinv[row]*dinv[col]
__global__ __launch_bounds__(256)
void scatter_add(const float* __restrict__ src, float* __restrict__ dst,
                 const long long* __restrict__ rows, const long long* __restrict__ cols,
                 const float* __restrict__ dinv, long long E, int D)
{
    int lane = threadIdx.x & 31;
    long long e = (long long)blockIdx.x * 8 + (threadIdx.x >> 5);
    if (e >= E) return;
    long long r = rows[e], c = cols[e];
    float norm = dinv[r] * dinv[c];
    const float* s = src + (size_t)r * D;
    float*       d = dst + (size_t)c * D;
    for (int f = lane; f < D; f += 32)
        atomicAdd(&d[f], s[f] * norm);
}

// ---------------------------------------------------------------------------
// Fused bias + LayerNorm (+ optional ReLU). One wave per row.
// ---------------------------------------------------------------------------
__global__ __launch_bounds__(256)
void bias_ln(const float* __restrict__ in, const float* __restrict__ bias,
             const float* __restrict__ w, const float* __restrict__ b,
             float* __restrict__ out, int n, int D, int relu)
{
    int lane = threadIdx.x & 31;
    int row  = blockIdx.x * 8 + (threadIdx.x >> 5);
    if (row >= n) return;
    const float* x = in + (size_t)row * D;
    const int per = D >> 5;                 // 8 (D=256) or 4 (D=128)
    float vals[8];
    float s = 0.f;
    #pragma unroll 8
    for (int i = 0; i < per; ++i) {
        int f = i * 32 + lane;
        vals[i] = x[f] + bias[f];
        s += vals[i];
    }
    #pragma unroll
    for (int m = 16; m > 0; m >>= 1) s += __shfl_xor(s, m, 32);
    float mu = s / (float)D;
    float vs = 0.f;
    #pragma unroll 8
    for (int i = 0; i < per; ++i) { float d = vals[i] - mu; vs += d * d; }
    #pragma unroll
    for (int m = 16; m > 0; m >>= 1) vs += __shfl_xor(vs, m, 32);
    float inv = rsqrtf(vs / (float)D + 1e-5f);
    float* o = out + (size_t)row * D;
    #pragma unroll 8
    for (int i = 0; i < per; ++i) {
        int f = i * 32 + lane;
        float v = (vals[i] - mu) * inv * w[f] + b[f];
        if (relu) v = fmaxf(v, 0.f);
        o[f] = v;
    }
}

// ---------------------------------------------------------------------------
extern "C" void kernel_launch(void* const* d_in, const int* in_sizes, int n_in,
                              void* d_out, int out_size, void* d_ws, size_t ws_size,
                              hipStream_t stream) {
    (void)n_in; (void)out_size; (void)ws_size;
    const float*     x    = (const float*)d_in[0];
    const long long* edge = (const long long*)d_in[1];   // int64 [2,E]
    const float*     W1   = (const float*)d_in[2];
    const float*     b1   = (const float*)d_in[3];
    const float*     ln1w = (const float*)d_in[4];
    const float*     ln1b = (const float*)d_in[5];
    const float*     W2   = (const float*)d_in[6];
    const float*     b2   = (const float*)d_in[7];
    const float*     ln2w = (const float*)d_in[8];
    const float*     ln2b = (const float*)d_in[9];

    const int n = in_sizes[0] / 512;                  // 50000
    const long long E = (long long)in_sizes[1] / 2;   // 800000
    const long long* rows = edge;                     // edge_index[0]
    const long long* cols = edge + E;                 // edge_index[1]

    float* ws   = (float*)d_ws;
    size_t nd1  = (size_t)n * 256;
    float* buf0 = ws;                 // xw1 -> h1 -> agg2
    float* buf1 = ws + nd1;           // agg1 -> xw2
    float* deg  = ws + 2 * nd1;
    float* dinv = deg + n;

    dim3 blk(256);

    // degree normalization
    deg_init  <<<(n + 255) / 256, blk, 0, stream>>>(deg, n);
    deg_count <<<(unsigned)((E + 255) / 256), blk, 0, stream>>>(cols, deg, E);
    deg_rsqrt <<<(n + 255) / 256, blk, 0, stream>>>(deg, dinv, n);

    // layer 1: xw1 = x @ W1
    wmma_gemm_f16<<<dim3((n + 127) / 128, 4), blk, 0, stream>>>(x, W1, buf0, n, 512, 256);
    self_init   <<<(unsigned)(((size_t)n * 256 + 255) / 256), blk, 0, stream>>>(buf0, dinv, buf1, n, 256);
    scatter_add <<<(unsigned)((E + 7) / 8), blk, 0, stream>>>(buf0, buf1, rows, cols, dinv, E, 256);
    bias_ln     <<<(n + 7) / 8, blk, 0, stream>>>(buf1, b1, ln1w, ln1b, buf0, n, 256, 1);

    // layer 2: xw2 = h1 @ W2
    wmma_gemm_f16<<<dim3((n + 127) / 128, 2), blk, 0, stream>>>(buf0, W2, buf1, n, 256, 128);
    self_init   <<<(unsigned)(((size_t)n * 128 + 255) / 256), blk, 0, stream>>>(buf1, dinv, buf0, n, 128);
    scatter_add <<<(unsigned)((E + 7) / 8), blk, 0, stream>>>(buf1, buf0, rows, cols, dinv, E, 128);
    bias_ln     <<<(n + 7) / 8, blk, 0, stream>>>(buf0, b2, ln2w, ln2b, (float*)d_out, n, 128, 0);
}
